// DistanceBiasedSelfAttention_19026705121718
// MI455X (gfx1250) — matrix operand order repaired and checked by reading, not compiled, not run on hardware
//
#include <hip/hip_runtime.h>
#include <stdint.h>

// ---------------- types / helpers ----------------
typedef __attribute__((ext_vector_type(16))) __bf16 v16bf;
typedef __attribute__((ext_vector_type(8)))  float  v8f;

static __device__ __forceinline__ uint32_t f2bf_bits(float f) {
  uint32_t u = __builtin_bit_cast(uint32_t, f);
  return (u + 0x7FFFu + ((u >> 16) & 1u)) >> 16;   // RNE f32 -> bf16
}
static __device__ __forceinline__ uint32_t packbf2(float lo, float hi) {
  return f2bf_bits(lo) | (f2bf_bits(hi) << 16);
}
struct U8 { uint32_t u[8]; };
static __device__ __forceinline__ v16bf mk_frag(uint4 a, uint4 b) {
  U8 t{{a.x, a.y, a.z, a.w, b.x, b.y, b.z, b.w}};
  return __builtin_bit_cast(v16bf, t);
}
static __device__ __forceinline__ v8f zero8() {
  v8f z = {0.f,0.f,0.f,0.f,0.f,0.f,0.f,0.f};
  return z;
}
#define BF16_WMMA(A, B, C) \
  __builtin_amdgcn_wmma_f32_16x16x32_bf16(false, (A), false, (B), (short)0, (C), false, false)

// Problem constants
#define BB 4
#define NN 2048
#define CC 1024
#define HH 16
#define DD 64
#define RR (BB * NN)   // 8192 rows

// ---------------- kernel 0: f32 -> bf16 pair pack ----------------
__global__ void cvt_bf16_kernel(const float2* __restrict__ src,
                                uint32_t* __restrict__ dst, int n) {
  int i = blockIdx.x * blockDim.x + threadIdx.x;
  if (i < n) { float2 v = src[i]; dst[i] = packbf2(v.x, v.y); }
}

// ---------------- kernel 1: fused QKV projection ----------------
// grid (RR/16, CC/64, 3), block 32. Each wave: 16x64 strip of x @ W^T + b.
__global__ __launch_bounds__(32) void qkv_gemm_kernel(
    const uint32_t* __restrict__ xb,
    const uint32_t* __restrict__ wq, const uint32_t* __restrict__ wk,
    const uint32_t* __restrict__ wv,
    const float* __restrict__ bq, const float* __restrict__ bk,
    const float* __restrict__ bv,
    uint16_t* __restrict__ Q, uint16_t* __restrict__ K,
    uint16_t* __restrict__ Vt) {
  const int lane = threadIdx.x;
  const int lo = lane & 15, hf = lane >> 4;
  const int mbase = blockIdx.x * 16;
  const int n64 = blockIdx.y;
  const int z = blockIdx.z;
  const uint32_t* W   = (z == 0) ? wq : (z == 1) ? wk : wv;
  const float*    bia = (z == 0) ? bq : (z == 1) ? bk : bv;

  const uint32_t* arow = xb + (size_t)(mbase + lo) * 512 + hf * 8;
  const uint32_t* br0 = W + (size_t)(n64 * 64 +  0 + lo) * 512 + hf * 8;
  const uint32_t* br1 = W + (size_t)(n64 * 64 + 16 + lo) * 512 + hf * 8;
  const uint32_t* br2 = W + (size_t)(n64 * 64 + 32 + lo) * 512 + hf * 8;
  const uint32_t* br3 = W + (size_t)(n64 * 64 + 48 + lo) * 512 + hf * 8;

  v8f acc[4] = {zero8(), zero8(), zero8(), zero8()};
#pragma unroll 4
  for (int ks = 0; ks < 32; ++ks) {
    v16bf a = mk_frag(*(const uint4*)(arow + ks * 16),
                      *(const uint4*)(arow + ks * 16 + 4));
    v16bf b;
    b = mk_frag(*(const uint4*)(br0 + ks * 16), *(const uint4*)(br0 + ks * 16 + 4));
    acc[0] = BF16_WMMA(a, b, acc[0]);
    b = mk_frag(*(const uint4*)(br1 + ks * 16), *(const uint4*)(br1 + ks * 16 + 4));
    acc[1] = BF16_WMMA(a, b, acc[1]);
    b = mk_frag(*(const uint4*)(br2 + ks * 16), *(const uint4*)(br2 + ks * 16 + 4));
    acc[2] = BF16_WMMA(a, b, acc[2]);
    b = mk_frag(*(const uint4*)(br3 + ks * 16), *(const uint4*)(br3 + ks * 16 + 4));
    acc[3] = BF16_WMMA(a, b, acc[3]);
  }

#pragma unroll
  for (int t = 0; t < 4; ++t) {
    const int c  = n64 * 64 + t * 16 + lo;
    const float bvv = bia[c];
    const int h = c >> 6, d = c & 63;
#pragma unroll
    for (int v = 0; v < 8; ++v) {
      const int r = mbase + v + 8 * hf;
      const int bi = r >> 11, q = r & (NN - 1);
      const uint16_t o = (uint16_t)f2bf_bits(acc[t][v] + bvv);
      if (z == 2)
        Vt[(((size_t)(bi * HH + h)) * DD + d) * NN + q] = o;
      else if (z == 0)
        Q[(((size_t)(bi * HH + h)) * NN + q) * DD + d] = o;
      else
        K[(((size_t)(bi * HH + h)) * NN + q) * DD + d] = o;
    }
  }
}

// ---------------- kernel 2: flash attention with distance bias ----------------
// grid (NN/16, HH, BB), block 32. One wave: 16 query rows, stream 2048 keys.
__global__ __launch_bounds__(32) void attn_kernel(
    const uint32_t* __restrict__ Qb, const uint32_t* __restrict__ Kb,
    const uint32_t* __restrict__ Vtb,
    const float* __restrict__ dist, const float* __restrict__ dwp,
    const float* __restrict__ dbp, uint16_t* __restrict__ Ob) {
  __shared__ __align__(16) uint16_t pls[16 * 32];

  const int lane = threadIdx.x;
  const int lo = lane & 15, hf = lane >> 4;
  const int qbase = blockIdx.x * 16;
  const int h = blockIdx.y, b = blockIdx.z;
  const float dw = dwp[0], db = dbp[0];
  const size_t headrow = ((size_t)(b * HH + h)) * NN;   // row offset (in rows)

  // Q A-fragments (persist across whole key loop)
  const uint32_t* qrow = Qb + (headrow + qbase + lo) * 32;
  const v16bf aq0 = mk_frag(*(const uint4*)(qrow + hf * 8),
                            *(const uint4*)(qrow + hf * 8 + 4));
  const v16bf aq1 = mk_frag(*(const uint4*)(qrow + 16 + hf * 8),
                            *(const uint4*)(qrow + 16 + hf * 8 + 4));

  const uint32_t* vrow[4];
#pragma unroll
  for (int t = 0; t < 4; ++t)
    vrow[t] = Vtb + (((size_t)(b * HH + h)) * DD + t * 16 + lo) * 1024 + hf * 8;

  float mr[8], lr[8];
  v8f acc[4] = {zero8(), zero8(), zero8(), zero8()};
#pragma unroll
  for (int v = 0; v < 8; ++v) { mr[v] = -3.0e38f; lr[v] = 0.f; }

  for (int kb = 0; kb < NN; kb += 32) {
    // --- S = Q K^T (two 16x16 tiles, K-depth 64 = 2 wmma each) ---
    const uint32_t* kr0 = Kb + (headrow + kb + lo) * 32;
    const uint32_t* kr1 = Kb + (headrow + kb + 16 + lo) * 32;
    v8f s0 = zero8(), s1 = zero8();
    v16bf bk;
    bk = mk_frag(*(const uint4*)(kr0 + hf * 8), *(const uint4*)(kr0 + hf * 8 + 4));
    s0 = BF16_WMMA(aq0, bk, s0);
    bk = mk_frag(*(const uint4*)(kr0 + 16 + hf * 8), *(const uint4*)(kr0 + 16 + hf * 8 + 4));
    s0 = BF16_WMMA(aq1, bk, s0);
    bk = mk_frag(*(const uint4*)(kr1 + hf * 8), *(const uint4*)(kr1 + hf * 8 + 4));
    s1 = BF16_WMMA(aq0, bk, s1);
    bk = mk_frag(*(const uint4*)(kr1 + 16 + hf * 8), *(const uint4*)(kr1 + 16 + hf * 8 + 4));
    s1 = BF16_WMMA(aq1, bk, s1);

    // --- scale + distance bias + online softmax (per row v) ---
#pragma unroll
    for (int v = 0; v < 8; ++v) {
      const int q = qbase + v + 8 * hf;
      const float* drow = dist + (size_t)q * NN + kb;
      float x0 = s0[v] * 0.125f + dw * drow[lo] + db;
      float x1 = s1[v] * 0.125f + dw * drow[16 + lo] + db;
      float rm = fmaxf(x0, x1);
      rm = fmaxf(rm, __shfl_xor(rm, 1, 32));
      rm = fmaxf(rm, __shfl_xor(rm, 2, 32));
      rm = fmaxf(rm, __shfl_xor(rm, 4, 32));
      rm = fmaxf(rm, __shfl_xor(rm, 8, 32));
      const float mn   = fmaxf(mr[v], rm);
      const float corr = __expf(mr[v] - mn);
      const float p0 = __expf(x0 - mn);
      const float p1 = __expf(x1 - mn);
      float rs = p0 + p1;
      rs += __shfl_xor(rs, 1, 32);
      rs += __shfl_xor(rs, 2, 32);
      rs += __shfl_xor(rs, 4, 32);
      rs += __shfl_xor(rs, 8, 32);
      lr[v] = lr[v] * corr + rs;
      mr[v] = mn;
      acc[0][v] *= corr; acc[1][v] *= corr;
      acc[2][v] *= corr; acc[3][v] *= corr;
      const int row = v + 8 * hf;
      pls[row * 32 + lo]      = (uint16_t)f2bf_bits(p0);
      pls[row * 32 + 16 + lo] = (uint16_t)f2bf_bits(p1);
    }

    // --- re-shape P (D-layout -> A-layout) through LDS ---
    __syncthreads();
    const uint32_t* pr = (const uint32_t*)pls + lo * 16;
    const v16bf ap = mk_frag(*(const uint4*)(pr + hf * 8),
                             *(const uint4*)(pr + hf * 8 + 4));
    __syncthreads();

    // --- O += P V (4 output d-tiles, K-depth 32 keys) ---
    const int kdw = kb >> 1;
#pragma unroll
    for (int t = 0; t < 4; ++t) {
      v16bf bv = mk_frag(*(const uint4*)(vrow[t] + kdw),
                         *(const uint4*)(vrow[t] + kdw + 4));
      acc[t] = BF16_WMMA(ap, bv, acc[t]);
    }
  }

  // --- normalize and emit O in [RR, CC] bf16 layout ---
#pragma unroll
  for (int v = 0; v < 8; ++v) {
    const float inv = 1.0f / lr[v];
    const size_t r = (size_t)(b * NN + qbase + v + 8 * hf);
#pragma unroll
    for (int t = 0; t < 4; ++t) {
      const int c = h * 64 + t * 16 + lo;
      Ob[r * CC + c] = (uint16_t)f2bf_bits(acc[t][v] * inv);
    }
  }
}

// ---------------- kernel 3: output projection O @ Wo^T + bo (f32 out) ----------------
__global__ __launch_bounds__(32) void out_gemm_kernel(
    const uint32_t* __restrict__ Ob, const uint32_t* __restrict__ wo,
    const float* __restrict__ bo, float* __restrict__ out) {
  const int lane = threadIdx.x;
  const int lo = lane & 15, hf = lane >> 4;
  const int mbase = blockIdx.x * 16;
  const int n64 = blockIdx.y;

  const uint32_t* arow = Ob + (size_t)(mbase + lo) * 512 + hf * 8;
  const uint32_t* br0 = wo + (size_t)(n64 * 64 +  0 + lo) * 512 + hf * 8;
  const uint32_t* br1 = wo + (size_t)(n64 * 64 + 16 + lo) * 512 + hf * 8;
  const uint32_t* br2 = wo + (size_t)(n64 * 64 + 32 + lo) * 512 + hf * 8;
  const uint32_t* br3 = wo + (size_t)(n64 * 64 + 48 + lo) * 512 + hf * 8;

  v8f acc[4] = {zero8(), zero8(), zero8(), zero8()};
#pragma unroll 4
  for (int ks = 0; ks < 32; ++ks) {
    v16bf a = mk_frag(*(const uint4*)(arow + ks * 16),
                      *(const uint4*)(arow + ks * 16 + 4));
    v16bf b;
    b = mk_frag(*(const uint4*)(br0 + ks * 16), *(const uint4*)(br0 + ks * 16 + 4));
    acc[0] = BF16_WMMA(a, b, acc[0]);
    b = mk_frag(*(const uint4*)(br1 + ks * 16), *(const uint4*)(br1 + ks * 16 + 4));
    acc[1] = BF16_WMMA(a, b, acc[1]);
    b = mk_frag(*(const uint4*)(br2 + ks * 16), *(const uint4*)(br2 + ks * 16 + 4));
    acc[2] = BF16_WMMA(a, b, acc[2]);
    b = mk_frag(*(const uint4*)(br3 + ks * 16), *(const uint4*)(br3 + ks * 16 + 4));
    acc[3] = BF16_WMMA(a, b, acc[3]);
  }

#pragma unroll
  for (int t = 0; t < 4; ++t) {
    const int c = n64 * 64 + t * 16 + lo;
    const float bb = bo[c];
#pragma unroll
    for (int v = 0; v < 8; ++v) {
      const size_t r = (size_t)(mbase + v + 8 * hf);
      out[r * CC + c] = acc[t][v] + bb;
    }
  }
}

// ---------------- host launcher ----------------
extern "C" void kernel_launch(void* const* d_in, const int* in_sizes, int n_in,
                              void* d_out, int out_size, void* d_ws, size_t ws_size,
                              hipStream_t stream) {
  const float* x    = (const float*)d_in[0];
  const float* dist = (const float*)d_in[1];
  const float* Wq   = (const float*)d_in[2];
  const float* bq   = (const float*)d_in[3];
  const float* Wk   = (const float*)d_in[4];
  const float* bk   = (const float*)d_in[5];
  const float* Wv   = (const float*)d_in[6];
  const float* bv   = (const float*)d_in[7];
  const float* Wo   = (const float*)d_in[8];
  const float* bo   = (const float*)d_in[9];
  const float* dwp  = (const float*)d_in[10];
  const float* dbp  = (const float*)d_in[11];
  float* out = (float*)d_out;

  // workspace layout (bf16 buffers)
  char* ws = (char*)d_ws;
  const size_t szXB = (size_t)RR * CC * 2;     // x in bf16
  const size_t szW  = (size_t)CC * CC * 2;     // one weight in bf16
  const size_t szH  = (size_t)BB * HH * NN * DD * 2;  // Q/K/Vt/O each
  uint32_t* xb  = (uint32_t*)(ws);
  uint32_t* wqb = (uint32_t*)(ws + szXB);
  uint32_t* wkb = (uint32_t*)(ws + szXB + szW);
  uint32_t* wvb = (uint32_t*)(ws + szXB + 2 * szW);
  uint32_t* wob = (uint32_t*)(ws + szXB + 3 * szW);
  uint16_t* Qb  = (uint16_t*)(ws + szXB + 4 * szW);
  uint16_t* Kb  = (uint16_t*)(ws + szXB + 4 * szW + szH);
  uint16_t* Vtb = (uint16_t*)(ws + szXB + 4 * szW + 2 * szH);
  uint16_t* Ob  = (uint16_t*)(ws + szXB + 4 * szW + 3 * szH);

  // 0) bf16 conversions
  {
    int nx = RR * CC / 2, nw = CC * CC / 2;
    cvt_bf16_kernel<<<(nx + 255) / 256, 256, 0, stream>>>((const float2*)x,  xb,  nx);
    cvt_bf16_kernel<<<(nw + 255) / 256, 256, 0, stream>>>((const float2*)Wq, wqb, nw);
    cvt_bf16_kernel<<<(nw + 255) / 256, 256, 0, stream>>>((const float2*)Wk, wkb, nw);
    cvt_bf16_kernel<<<(nw + 255) / 256, 256, 0, stream>>>((const float2*)Wv, wvb, nw);
    cvt_bf16_kernel<<<(nw + 255) / 256, 256, 0, stream>>>((const float2*)Wo, wob, nw);
  }
  // 1) fused QKV projection
  qkv_gemm_kernel<<<dim3(RR / 16, CC / 64, 3), 32, 0, stream>>>(
      xb, wqb, wkb, wvb, bq, bk, bv, Qb, Kb, Vtb);
  // 2) flash attention with distance bias
  attn_kernel<<<dim3(NN / 16, HH, BB), 32, 0, stream>>>(
      (const uint32_t*)Qb, (const uint32_t*)Kb, (const uint32_t*)Vtb,
      dist, dwp, dbp, Ob);
  // 3) output projection
  out_gemm_kernel<<<dim3(RR / 16, CC / 64), 32, 0, stream>>>(
      (const uint32_t*)Ob, wob, bo, out);
}
